// SelfAttention3D_78417512890690
// MI455X (gfx1250) — matrix operand order repaired and checked by reading, not compiled, not run on hardware
//
#include <hip/hip_runtime.h>

// Self-attention over 3D volume, split-K flash attention, bf16 WMMA on gfx1250.
// B=4, C=256, CR=32, N=4096.

#define BN    4
#define CH    256
#define CRD   32
#define NN    4096
#define SPLIT 4                  // key-range splits for occupancy
#define JCH   (NN / 32 / SPLIT)  // 32 key-chunk iterations per split
#define VSTR  40                 // padded LDS stride (halfs) for a 32-half V row

#if __has_builtin(__builtin_amdgcn_global_load_async_to_lds_b128)
#define HAVE_ASYNC 1
#else
#define HAVE_ASYNC 0
#endif

#if HAVE_ASYNC
#if __has_builtin(__builtin_amdgcn_s_wait_asynccnt)
#define WAIT_ASYNC(n) __builtin_amdgcn_s_wait_asynccnt(n)
#else
#define WAIT_ASYNC(n) asm volatile("s_wait_asynccnt %0" ::"i"(n) : "memory")
#endif
#else
#define WAIT_ASYNC(n) do {} while (0)
#endif

typedef __attribute__((ext_vector_type(16))) __bf16 v16bf;
typedef __attribute__((ext_vector_type(8)))  __bf16 v8bf;
typedef __attribute__((ext_vector_type(8)))  float  v8f;
typedef __attribute__((ext_vector_type(4)))  int    v4i;
// async-load-to-LDS pointer types (per clang diagnostic: int4 in AS1 / AS3)
typedef __attribute__((address_space(1))) v4i gv4i;
typedef __attribute__((address_space(3))) v4i lv4i;

static __device__ __forceinline__ v16bf cat8(v8bf lo, v8bf hi) {
  return __builtin_shufflevector(lo, hi, 0,1,2,3,4,5,6,7,8,9,10,11,12,13,14,15);
}

static __device__ __forceinline__ v8bf cvt8f(const float* __restrict__ p) {
  v8bf r;
#pragma unroll
  for (int i = 0; i < 8; ++i) r[i] = (__bf16)p[i];
  return r;
}

static __device__ __forceinline__ v8f wmma_bf16(v16bf a, v16bf b, v8f c) {
  return __builtin_amdgcn_wmma_f32_16x16x32_bf16(false, a, false, b, (short)0, c,
                                                 false, false);
}

// ---------------------------------------------------------------------------
// Kernel 1: Q/K projection.  Q[n][cr] = sum_c Wq[cr][c] * x[b][c][n]  (bf16)
// ---------------------------------------------------------------------------
__global__ void __launch_bounds__(32)
proj_qk_kernel(const float* __restrict__ x, const float* __restrict__ Wq,
               const float* __restrict__ Wk, __bf16* __restrict__ Qb,
               __bf16* __restrict__ Kb) {
  const int lane = threadIdx.x & 31;
  const int r = lane & 15, g = lane >> 4;
  const int b  = blockIdx.x / (NN / 16);
  const int n0 = (blockIdx.x % (NN / 16)) * 16;
  const int mat = blockIdx.y >> 1;
  const int cr0 = (blockIdx.y & 1) * 16;
  const float* __restrict__ W = mat ? Wk : Wq;
  __bf16* __restrict__ dst    = mat ? Kb : Qb;

  v8f acc = {0.f, 0.f, 0.f, 0.f, 0.f, 0.f, 0.f, 0.f};
  for (int k0 = 0; k0 < CH; k0 += 32) {
    const float* wrow = W + (size_t)(cr0 + r) * CH + k0;
    v16bf a = cat8(cvt8f(wrow + 16 * g), cvt8f(wrow + 16 * g + 8));
    const float* xp = x + ((size_t)b * CH + k0 + 16 * g) * NN + n0 + r;
    v16bf bx;
#pragma unroll
    for (int e = 0; e < 16; ++e) bx[e] = (__bf16)xp[(size_t)e * NN];
    acc = wmma_bf16(a, bx, acc);
  }
  v8bf o;
#pragma unroll
  for (int v = 0; v < 8; ++v) o[v] = (__bf16)acc[v];
  *(v8bf*)(dst + ((size_t)b * NN + n0 + r) * CRD + cr0 + 8 * g) = o;
}

// ---------------------------------------------------------------------------
// Kernel 2: V projection.  Vb[b][c_out][n] (bf16, x-like layout).
// ---------------------------------------------------------------------------
__global__ void __launch_bounds__(32)
proj_v_kernel(const float* __restrict__ x, const float* __restrict__ Wv,
              __bf16* __restrict__ Vb) {
  const int lane = threadIdx.x & 31;
  const int r = lane & 15, g = lane >> 4;
  const int b  = blockIdx.x / (NN / 16);
  const int n0 = (blockIdx.x % (NN / 16)) * 16;
  const int c0 = blockIdx.y * 16;

  v8f acc = {0.f, 0.f, 0.f, 0.f, 0.f, 0.f, 0.f, 0.f};
  for (int k0 = 0; k0 < CH; k0 += 32) {
    const float* wrow = Wv + (size_t)(c0 + r) * CH + k0;
    v16bf a = cat8(cvt8f(wrow + 16 * g), cvt8f(wrow + 16 * g + 8));
    const float* xp = x + ((size_t)b * CH + k0 + 16 * g) * NN + n0 + r;
    v16bf bx;
#pragma unroll
    for (int e = 0; e < 16; ++e) bx[e] = (__bf16)xp[(size_t)e * NN];
    acc = wmma_bf16(a, bx, acc);
  }
#pragma unroll
  for (int v = 0; v < 8; ++v)
    Vb[((size_t)b * CH + c0 + v + 8 * g) * NN + n0 + r] = (__bf16)acc[v];
}

// ---------------------------------------------------------------------------
// Kernel 3: split-K flash attention.  256-thread block = 8 waves = 128 query
// rows, one key-range split per block.  Each 32x256 V chunk is staged in LDS
// once per block (async double-buffered), shared by all 8 waves.  Partials
// are unnormalized O plus per-row (m, l) for the log-sum-exp combine.
// ---------------------------------------------------------------------------
__global__ void __launch_bounds__(256)
flash_kernel(const __bf16* __restrict__ Qb, const __bf16* __restrict__ Kb,
             const __bf16* __restrict__ Vb, __bf16* __restrict__ Opart,
             float* __restrict__ Ml) {
  __shared__ __bf16 v_lds[2][256 * VSTR];  // 40 KB: V chunk double buffer
  __shared__ __bf16 p_lds[8][16 * 32];     //  8 KB: per-wave P transpose

  const int tid  = threadIdx.x;
  const int w    = tid >> 5;
  const int lane = tid & 31;
  const int r = lane & 15, g = lane >> 4;
  const int b      = blockIdx.x / (NN / 128);
  const int rowblk = blockIdx.x % (NN / 128);
  const int s      = blockIdx.y;
  const int n0     = rowblk * 128 + w * 16;
  const int jstart = s * JCH;
  const float sc = 0.17677669529663687f;  // 1/sqrt(CR)

  const __bf16* kbase = Kb + (size_t)b * NN * CRD;
  const __bf16* vrow  = Vb + (size_t)(b * CH + tid) * NN;  // this thread's V row

  // Stage 32 keys' worth of V (32 x 256 bf16) for chunk j2 into buffer buf:
  // thread tid copies its channel row (64 B) as 4 x b128.
  auto stage = [&](int j2, int buf) {
    const __bf16* src = vrow + j2 * 32;
#pragma unroll
    for (int seg = 0; seg < 4; ++seg) {
#if HAVE_ASYNC
      __builtin_amdgcn_global_load_async_to_lds_b128(
          (gv4i*)(src + seg * 8),
          (lv4i*)(&v_lds[buf][tid * VSTR + seg * 8]),
          0, 0);
#else
      *(uint4*)(&v_lds[buf][tid * VSTR + seg * 8]) =
          *(const uint4*)(src + seg * 8);
#endif
    }
  };

  // Q A-operand: row = r, K(e,g) = 16g+e.
  const __bf16* qrow = Qb + ((size_t)b * NN + n0 + r) * CRD;
  const v16bf aQ = cat8(*(const v8bf*)(qrow + 16 * g),
                        *(const v8bf*)(qrow + 16 * g + 8));

  float m[8], lsum[8];
  v8f o_acc[16];
  const v8f vzero = {0.f, 0.f, 0.f, 0.f, 0.f, 0.f, 0.f, 0.f};
#pragma unroll
  for (int v = 0; v < 8; ++v) { m[v] = -1e30f; lsum[v] = 0.f; }
#pragma unroll
  for (int t = 0; t < 16; ++t) o_acc[t] = vzero;

  stage(jstart, 0);  // prologue: fill buffer 0

  for (int it = 0; it < JCH; ++it) {
    const int j2  = jstart + it;
    const int buf = it & 1;
    const bool more = (it + 1) < JCH;
    if (more) stage(j2 + 1, buf ^ 1);

    const int nj0 = j2 * 32;
    v16bf bK0 = *(const v16bf*)(kbase + (size_t)(nj0 + r) * CRD + 16 * g);
    v16bf bK1 = *(const v16bf*)(kbase + (size_t)(nj0 + 16 + r) * CRD + 16 * g);
    if (more)
      __builtin_prefetch(kbase + (size_t)(nj0 + 32 + r) * CRD, 0, 0);

    v8f s0 = wmma_bf16(aQ, bK0, vzero);
    v8f s1 = wmma_bf16(aQ, bK1, vzero);

    // Online softmax; S layout: row M=v+8g, col=lane%16 -> butterfly shuffles
    // within each 16-lane half reduce along columns.
    float alpha[8], p0[8], p1[8];
#pragma unroll
    for (int v = 0; v < 8; ++v) {
      float a0 = s0[v] * sc, a1 = s1[v] * sc;
      float t = fmaxf(a0, a1);
      t = fmaxf(t, __shfl_xor(t, 1, 32));
      t = fmaxf(t, __shfl_xor(t, 2, 32));
      t = fmaxf(t, __shfl_xor(t, 4, 32));
      t = fmaxf(t, __shfl_xor(t, 8, 32));
      float mn = fmaxf(m[v], t);
      alpha[v] = __expf(m[v] - mn);
      m[v] = mn;
      float e0 = __expf(a0 - mn), e1 = __expf(a1 - mn);
      p0[v] = e0; p1[v] = e1;
      float rs = e0 + e1;
      rs += __shfl_xor(rs, 1, 32);
      rs += __shfl_xor(rs, 2, 32);
      rs += __shfl_xor(rs, 4, 32);
      rs += __shfl_xor(rs, 8, 32);
      lsum[v] = lsum[v] * alpha[v] + rs;
    }
#pragma unroll
    for (int t = 0; t < 16; ++t)
#pragma unroll
      for (int v = 0; v < 8; ++v) o_acc[t][v] *= alpha[v];

    // Transpose P (C/D layout) -> A layout through per-wave LDS.
#pragma unroll
    for (int v = 0; v < 8; ++v) {
      p_lds[w][(v + 8 * g) * 32 + r]      = (__bf16)p0[v];
      p_lds[w][(v + 8 * g) * 32 + 16 + r] = (__bf16)p1[v];
    }

    if (more) { WAIT_ASYNC(4); } else { WAIT_ASYNC(0); }
    __syncthreads();  // V chunk `buf` visible to all waves; P staged

    const __bf16* prow = &p_lds[w][r * 32];
    v16bf aP = cat8(*(const v8bf*)(prow + 16 * g),
                    *(const v8bf*)(prow + 16 * g + 8));

    // O[t] += P(16x32) * V(32x16), V B-operand from LDS.
#pragma unroll
    for (int t = 0; t < 16; ++t) {
      const __bf16* vp = &v_lds[buf][(t * 16 + r) * VSTR + 16 * g];
      v16bf bV = cat8(*(const v8bf*)(vp), *(const v8bf*)(vp + 8));
      o_acc[t] = wmma_bf16(aP, bV, o_acc[t]);
    }
    __syncthreads();  // done with buf / p_lds before next overwrite
  }

  // Store unnormalized partial O (bf16) + per-row (m, l) for combine.
#pragma unroll
  for (int t = 0; t < 16; ++t)
#pragma unroll
    for (int v = 0; v < 8; ++v)
      Opart[((size_t)(b * SPLIT + s) * NN + n0 + v + 8 * g) * CH + t * 16 + r] =
          (__bf16)o_acc[t][v];
  if (r == 0) {
#pragma unroll
    for (int v = 0; v < 8; ++v) {
      size_t mi = ((size_t)(b * SPLIT + s) * NN + n0 + v + 8 * g) * 2;
      Ml[mi]     = m[v];
      Ml[mi + 1] = lsum[v];
    }
  }
}

// ---------------------------------------------------------------------------
// Kernel 4: log-sum-exp combine of the SPLIT partials -> O[b][n][c] f32.
// ---------------------------------------------------------------------------
__global__ void __launch_bounds__(256)
combine_kernel(const __bf16* __restrict__ Opart, const float* __restrict__ Ml,
               float* __restrict__ O) {
  const size_t idx = (size_t)blockIdx.x * 256 + threadIdx.x;  // over B*N*C
  const int c = (int)(idx % CH);
  const int n = (int)((idx / CH) % NN);
  const int b = (int)(idx / ((size_t)CH * NN));

  float mv[SPLIT], lv[SPLIT];
  float M = -1e30f;
#pragma unroll
  for (int s = 0; s < SPLIT; ++s) {
    size_t mi = ((size_t)(b * SPLIT + s) * NN + n) * 2;
    mv[s] = Ml[mi];
    lv[s] = Ml[mi + 1];
    M = fmaxf(M, mv[s]);
  }
  float denom = 0.f, acc = 0.f;
#pragma unroll
  for (int s = 0; s < SPLIT; ++s) {
    float wgt = __expf(mv[s] - M);
    denom += wgt * lv[s];
    acc += wgt * (float)Opart[((size_t)(b * SPLIT + s) * NN + n) * CH + c];
  }
  O[idx] = acc / denom;
}

// ---------------------------------------------------------------------------
// Kernel 5: output projection + residual.  y = gamma * (Wo @ O^T) + x.
// ---------------------------------------------------------------------------
__global__ void __launch_bounds__(32)
proj_o_kernel(const float* __restrict__ x, const float* __restrict__ Wo,
              const float* __restrict__ gamma, const float* __restrict__ Oin,
              float* __restrict__ y) {
  const int lane = threadIdx.x & 31;
  const int r = lane & 15, g = lane >> 4;
  const int b  = blockIdx.x / (CH / 16);
  const int o0 = (blockIdx.x % (CH / 16)) * 16;
  const int n0 = blockIdx.y * 16;

  v8f acc = {0.f, 0.f, 0.f, 0.f, 0.f, 0.f, 0.f, 0.f};
  for (int k0 = 0; k0 < CH; k0 += 32) {
    const float* wrow = Wo + (size_t)(o0 + r) * CH + k0;
    v16bf a = cat8(cvt8f(wrow + 16 * g), cvt8f(wrow + 16 * g + 8));
    const float* orow = Oin + ((size_t)b * NN + n0 + r) * CH + k0 + 16 * g;
    v16bf bo = cat8(cvt8f(orow), cvt8f(orow + 8));
    acc = wmma_bf16(a, bo, acc);
  }
  const float gm = gamma[0];
#pragma unroll
  for (int v = 0; v < 8; ++v) {
    size_t idx = ((size_t)b * CH + o0 + v + 8 * g) * NN + n0 + r;
    y[idx] = gm * acc[v] + x[idx];
  }
}

// ---------------------------------------------------------------------------
extern "C" void kernel_launch(void* const* d_in, const int* in_sizes, int n_in,
                              void* d_out, int out_size, void* d_ws,
                              size_t ws_size, hipStream_t stream) {
  (void)in_sizes; (void)n_in; (void)out_size; (void)ws_size;
  const float* x     = (const float*)d_in[0];
  const float* Wq    = (const float*)d_in[1];
  const float* Wk    = (const float*)d_in[2];
  const float* Wv    = (const float*)d_in[3];
  const float* Wo    = (const float*)d_in[4];
  const float* gamma = (const float*)d_in[5];
  float* out = (float*)d_out;

  char* ws = (char*)d_ws;
  __bf16* Qb = (__bf16*)(ws);               //  1 MB: [B][N][CR] bf16
  __bf16* Kb = (__bf16*)(ws + (1u << 20));  //  1 MB: [B][N][CR] bf16
  __bf16* Vb = (__bf16*)(ws + (2u << 20));  //  8 MB: [B][C][N]  bf16
  __bf16* Op = (__bf16*)(ws + (10u << 20)); // 32 MB: [B][S][N][C] bf16 partials
  float*  Ml = (float*)(ws + (42u << 20));  // .5 MB: [B][S][N][2] (m, l)
  float*  Ow = (float*)(ws + (43u << 20));  // 16 MB: [B][N][C]  f32 combined

  proj_qk_kernel<<<dim3(BN * NN / 16, 4), 32, 0, stream>>>(x, Wq, Wk, Qb, Kb);
  proj_v_kernel<<<dim3(BN * NN / 16, CH / 16), 32, 0, stream>>>(x, Wv, Vb);
  flash_kernel<<<dim3(BN * (NN / 128), SPLIT), 256, 0, stream>>>(Qb, Kb, Vb,
                                                                 Op, Ml);
  combine_kernel<<<dim3(BN * NN * CH / 256), 256, 0, stream>>>(Op, Ml, Ow);
  proj_o_kernel<<<dim3(BN * (CH / 16), NN / 16), 32, 0, stream>>>(x, Wo, gamma,
                                                                  Ow, out);
}
